// CAMM_18820546691539
// MI455X (gfx1250) — compile-verified, matrix operand order checked by reading, not run on hardware
//
#include <hip/hip_runtime.h>
#include <cstdint>

typedef __attribute__((ext_vector_type(16))) __bf16 bf16x16;
typedef __attribute__((ext_vector_type(8)))  float  f32x8;

#define HWPIX 65536
#define WIMG  256

__device__ __forceinline__ f32x8 wmma_bf16(bf16x16 a, bf16x16 b, f32x8 c) {
  return __builtin_amdgcn_wmma_f32_16x16x32_bf16(false, a, false, b, (short)0, c, false, false);
}

// 16-bit WMMA operand layout (ISA 7.12.2): per lane, vector elements 0-7 hold
// k = base..base+7 and elements 8-15 hold k = base+16..base+23 (base = kb+8*half).
// -> each fragment is exactly two 16-byte contiguous loads.
struct alignas(32) FragBits { uint4 a, b; };
__device__ __forceinline__ bf16x16 load_frag(const __bf16* p) {
  FragBits fb;
  fb.a = *(const uint4*)(p);        // k = base .. base+7
  fb.b = *(const uint4*)(p + 16);   // k = base+16 .. base+23
  return __builtin_bit_cast(bf16x16, fb);
}

union Pack8 { __bf16 h[8]; uint4 v; };

// ---------------------------------------------------------------------------
// Convert the four 128x128 weight matrices to bf16 (q_w, k_w, v_w, out_w)
// ---------------------------------------------------------------------------
__global__ __launch_bounds__(256) void CAMM_wcvt(const float* __restrict__ qw,
                                                 const float* __restrict__ kw,
                                                 const float* __restrict__ vw,
                                                 const float* __restrict__ ow,
                                                 __bf16* __restrict__ dst) {
  int i = blockIdx.x * 256 + threadIdx.x;   // < 65536
  int m = i >> 14, o = i & 16383;
  const float* s = (m == 0) ? qw : (m == 1) ? kw : (m == 2) ? vw : ow;
  dst[i] = (__bf16)s[o];
}

// ---------------------------------------------------------------------------
// Dense 1x1-conv GEMM: out[b,m,pix] = sum_k W[m,k]*in[b,k,pix] + bias[m]
// One block = 128 out-ch x 128 pixels, 8 waves, WMMA bf16.
// LDS tile is pixel-major [pix][k] so B-fragments are contiguous b128 loads.
// ---------------------------------------------------------------------------
__global__ __launch_bounds__(256) void CAMM_gemm128(const float* __restrict__ in,
                                                    const __bf16* __restrict__ wbf,
                                                    const float* __restrict__ bias,
                                                    float* __restrict__ out) {
  __shared__ alignas(16) __bf16 sIn[128 * 136];     // [pixel][k], 272B rows (16B aligned)
  const int pb = HWPIX / 128;                       // 512 pixel-tiles per batch
  int blk = blockIdx.x;
  int b = blk / pb;
  int base = b * 128 * HWPIX + (blk % pb) * 128;
  int tid = threadIdx.x;
  for (int e = tid; e < 2048; e += 256) {           // 8 channels per element -> 1 ds_store_b128
    int p = e & 127, k0 = (e >> 7) * 8;
    Pack8 u;
#pragma unroll
    for (int j = 0; j < 8; ++j) u.h[j] = (__bf16)in[base + (k0 + j) * HWPIX + p];
    *(uint4*)(sIn + p * 136 + k0) = u.v;
  }
  __syncthreads();
  int wv = tid >> 5, lane = tid & 31, hf = lane >> 4, lr = lane & 15;
  int mtile = wv;
  f32x8 acc[8];
  for (int t = 0; t < 8; ++t) acc[t] = (f32x8)(0.0f);
#pragma unroll
  for (int kb = 0; kb < 128; kb += 32) {
    bf16x16 af = load_frag(wbf + (mtile * 16 + lr) * 128 + kb + hf * 8);
#pragma unroll
    for (int nt = 0; nt < 8; ++nt) {
      bf16x16 bfr = load_frag(sIn + (nt * 16 + lr) * 136 + kb + hf * 8);
      acc[nt] = wmma_bf16(af, bfr, acc[nt]);
    }
  }
#pragma unroll
  for (int nt = 0; nt < 8; ++nt)
#pragma unroll
    for (int r = 0; r < 8; ++r) {
      int m = mtile * 16 + r + hf * 8;
      out[base + m * HWPIX + nt * 16 + lr] = acc[nt][r] + bias[m];
    }
}

// ---------------------------------------------------------------------------
// Fused predictor: t = leaky(LN(conv1x1([v,cg,cond_wind]))) kept in registers.
// ---------------------------------------------------------------------------
__global__ __launch_bounds__(256) void CAMM_predictor(
    const float* __restrict__ v, const float* __restrict__ cg,
    const float* __restrict__ p_in_w, const float* __restrict__ p_in_b,
    const float* __restrict__ ln_g, const float* __restrict__ ln_b,
    const float* __restrict__ conv_w, const float* __restrict__ conv_b,
    const float* __restrict__ off1_w, const float* __restrict__ off1_b,
    const float* __restrict__ off2_w, const float* __restrict__ off2_b,
    float* __restrict__ tmpix, float* __restrict__ x2,
    float* __restrict__ offx, float* __restrict__ offy, float* __restrict__ part) {
  __shared__ float sW[4356], sB[33], sG[33], sLB[33], sB2[33];
  __shared__ float sW2[1089], sO1[528], sO1b[16], sO2[32], sO2b[2];
  __shared__ float red[256];
  int tid = threadIdx.x;
  for (int i = tid; i < 4356; i += 256) sW[i] = p_in_w[i];
  for (int i = tid; i < 1089; i += 256) sW2[i] = conv_w[33 * 33 + i];
  for (int i = tid; i < 528; i += 256) sO1[i] = off1_w[i];
  if (tid < 33) { sB[tid] = p_in_b[tid]; sG[tid] = ln_g[tid]; sLB[tid] = ln_b[tid]; sB2[tid] = conv_b[33 + tid]; }
  if (tid < 16) sO1b[tid] = off1_b[tid];
  if (tid < 32) sO2[tid] = off2_w[tid];
  if (tid < 2)  sO2b[tid] = off2_b[tid];
  __syncthreads();

  int g = blockIdx.x * 256 + tid;          // < B*HW
  int b = g >> 16, pix = g & 65535;
  int y = pix >> 8, x = pix & 255;
  float cwy = -1.f + 2.f * (float)(y & 7) / 7.f;
  float cwx = -1.f + 2.f * (float)(x & 7) / 7.f;
  float cg0 = cg[b * 2 * HWPIX + pix], cg1 = cg[b * 2 * HWPIX + HWPIX + pix];

  float t[33];
#pragma unroll
  for (int ch = 0; ch < 33; ++ch)
    t[ch] = sB[ch] + sW[ch * 132 + 128] * cg0 + sW[ch * 132 + 129] * cg1 +
            sW[ch * 132 + 130] * cwy + sW[ch * 132 + 131] * cwx;
  int vbase = b * 128 * HWPIX + pix;
  for (int k = 0; k < 128; ++k) {
    float vk = v[vbase + k * HWPIX];
#pragma unroll
    for (int ch = 0; ch < 33; ++ch) t[ch] += sW[ch * 132 + k] * vk;
  }
  float mu = 0.f;
#pragma unroll
  for (int ch = 0; ch < 33; ++ch) mu += t[ch];
  mu /= 33.f;
  float var = 0.f;
#pragma unroll
  for (int ch = 0; ch < 33; ++ch) { float d = t[ch] - mu; var += d * d; }
  var /= 33.f;
  float inv = rsqrtf(var + 1e-6f);
#pragma unroll
  for (int ch = 0; ch < 33; ++ch) {
    float tt = sG[ch] * (t[ch] - mu) * inv + sLB[ch];
    t[ch] = tt > 0.f ? tt : 0.1f * tt;
  }
  float tm = 0.f;
#pragma unroll
  for (int ch = 0; ch < 33; ++ch) tm += t[ch];
  tmpix[g] = tm / 33.f;
  int xbase2 = b * 33 * HWPIX + pix;
  for (int o = 0; o < 33; ++o) {
    float a = sB2[o];
#pragma unroll
    for (int j = 0; j < 33; ++j) a += sW2[o * 33 + j] * t[j];
    x2[xbase2 + o * HWPIX] = a;
  }
  float h[16];
#pragma unroll
  for (int o = 0; o < 16; ++o) {
    float a = sO1b[o];
#pragma unroll
    for (int j = 0; j < 33; ++j) a += sO1[o * 33 + j] * t[j];
    h[o] = a > 0.f ? a : 0.1f * a;
  }
  float o0 = sO2b[0], o1 = sO2b[1];
#pragma unroll
  for (int j = 0; j < 16; ++j) { o0 += sO2[j] * h[j]; o1 += sO2[16 + j] * h[j]; }
  offx[g] = tanhf(o0) * 8.f;
  offy[g] = tanhf(o1) * 8.f;
  for (int ch = 0; ch < 33; ++ch) {
    red[tid] = t[ch];
    __syncthreads();
    for (int s = 128; s > 0; s >>= 1) { if (tid < s) red[tid] += red[tid + s]; __syncthreads(); }
    if (tid == 0) part[blockIdx.x * 33 + ch] = red[0];
    __syncthreads();
  }
}

__global__ void CAMM_tmean_fin(const float* __restrict__ part, float* __restrict__ tmean) {
  int tid = threadIdx.x;
  if (tid < 66) {
    int b = tid / 33, ch = tid % 33;
    float s = 0.f;
    for (int blk = b * 256; blk < (b + 1) * 256; ++blk) s += part[blk * 33 + ch];
    tmean[tid] = s / (float)HWPIX;
  }
}

__global__ void CAMM_ca(const float* __restrict__ tmean, const float* __restrict__ convw,
                        const float* __restrict__ convb, const float* __restrict__ caw,
                        const float* __restrict__ cab, float* __restrict__ ca) {
  __shared__ float tmp[33];
  int b = blockIdx.x, tid = threadIdx.x;
  if (tid < 33) {
    float a = convb[tid];
    for (int j = 0; j < 33; ++j) a += convw[tid * 33 + j] * tmean[b * 33 + j];
    tmp[tid] = a;
  }
  __syncthreads();
  float a = cab[tid];
  for (int q = 0; q < 33; ++q) a += caw[tid * 33 + q] * tmp[q];
  ca[b * 128 + tid] = 1.f / (1.f + expf(-a));
}

__global__ __launch_bounds__(256) void CAMM_sa(const float* __restrict__ x2,
                                               const float* __restrict__ saw,
                                               const float* __restrict__ sab,
                                               float* __restrict__ sa) {
  __shared__ float sW[297];
  int tid = threadIdx.x;
  for (int i = tid; i < 297; i += 256) sW[i] = saw[i];
  __syncthreads();
  int g = blockIdx.x * 256 + tid;
  int b = g >> 16, pix = g & 65535;
  int y = pix >> 8, x = pix & 255;
  float acc = sab[0];
  int base = b * 33 * HWPIX;
  for (int ch = 0; ch < 33; ++ch)
    for (int ky = 0; ky < 3; ++ky) {
      int yy = y + ky - 1;
      if (yy < 0 || yy > 255) continue;
      for (int kx = 0; kx < 3; ++kx) {
        int xx = x + kx - 1;
        if (xx < 0 || xx > 255) continue;
        acc += x2[base + ch * HWPIX + yy * WIMG + xx] * sW[ch * 9 + ky * 3 + kx];
      }
    }
  sa[g] = 1.f / (1.f + expf(-acc));
}

// ps = softmax over width of conv(conv(window-rearranged tmean)); one block per (b,hy)
__global__ void CAMM_ps(const float* __restrict__ tmpix, const float* __restrict__ w1,
                        const float* __restrict__ b1, const float* __restrict__ w2,
                        const float* __restrict__ b2, float* __restrict__ ps) {
  __shared__ float buf[64];
  int b = blockIdx.x >> 5, hy = blockIdx.x & 31, wx = threadIdx.x;
  float a[4] = {b1[0], b1[1], b1[2], b1[3]};
  for (int s = 0; s < 64; ++s) {
    int dh = s >> 3, dw = s & 7;
    float tv = tmpix[b * HWPIX + (hy * 8 + dh) * WIMG + wx * 8 + dw];
#pragma unroll
    for (int q = 0; q < 4; ++q) a[q] += w1[q * 64 + s] * tv;
  }
  float p0 = b2[0], p1 = b2[1];
#pragma unroll
  for (int q = 0; q < 4; ++q) { p0 += w2[q] * a[q]; p1 += w2[4 + q] * a[q]; }
  buf[wx] = p0; buf[32 + wx] = p1;
  __syncthreads();
  if (wx < 2) {
    float mx = -1e30f;
    for (int i = 0; i < 32; ++i) mx = fmaxf(mx, buf[wx * 32 + i]);
    float sm = 0.f;
    for (int i = 0; i < 32; ++i) sm += expf(buf[wx * 32 + i] - mx);
    for (int i = 0; i < 32; ++i)
      ps[b * 2048 + wx * 1024 + hy * 32 + i] = expf(buf[wx * 32 + i] - mx) / sm;
  }
}

__global__ void CAMM_mask(const float* __restrict__ ps, const float* __restrict__ gn,
                          float* __restrict__ mask) {
  int idx = blockIdx.x * 256 + threadIdx.x;   // < 2048
  int b = idx >> 10, win = idx & 1023;
  float v0 = ps[b * 2048 + 2 * win] + gn[idx * 2];
  float v1 = ps[b * 2048 + 2 * win + 1] + gn[idx * 2 + 1];
  mask[idx] = (v0 >= v1) ? 1.f : 0.f;
}

// k_img = x + bilinear flow_warp(x, offsets), border clamp, align_corners
__global__ __launch_bounds__(256) void CAMM_warp(const float* __restrict__ xin,
                                                 const float* __restrict__ offx,
                                                 const float* __restrict__ offy,
                                                 float* __restrict__ kimg) {
  int g = blockIdx.x * 256 + threadIdx.x;
  int b = g >> 16, pix = g & 65535;
  int y = pix >> 8, x = pix & 255;
  float px = fminf(fmaxf((float)x + offx[g], 0.f), 255.f);
  float py = fminf(fmaxf((float)y + offy[g], 0.f), 255.f);
  float x0 = floorf(px), y0 = floorf(py);
  float wxr = px - x0, wyr = py - y0;
  int x0i = (int)x0, y0i = (int)y0;
  int x1i = min(x0i + 1, 255), y1i = min(y0i + 1, 255);
  int base = b * 128 * HWPIX;
  for (int c = 0; c < 128; ++c) {
    int row = base + c * HWPIX;
    float v00 = xin[row + y0i * WIMG + x0i];
    float v01 = xin[row + y0i * WIMG + x1i];
    float v10 = xin[row + y1i * WIMG + x0i];
    float v11 = xin[row + y1i * WIMG + x1i];
    kimg[row + pix] = xin[row + pix] + v00 * (1.f - wxr) * (1.f - wyr) + v01 * wxr * (1.f - wyr) +
                      v10 * (1.f - wxr) * wyr + v11 * wxr * wyr;
  }
}

// ---------------------------------------------------------------------------
// Per-window attention. One block per window, 8 waves, WMMA for q1/k1.
// ---------------------------------------------------------------------------
#define SABS 136   // bf16 row stride for x/k window tile (272B, 16B aligned)
#define SQKS 132   // f32 row stride for q1/k1

__device__ __forceinline__ void win_load_tile(__bf16* sAB, const float* __restrict__ src,
                                              int xbase, int py0, int px0, int tid) {
  for (int e = tid; e < 1024; e += 256) {     // 8 channels -> 1 ds_store_b128
    int c0 = (e >> 6) * 8, s = e & 63;
    int gp = xbase + (py0 + (s >> 3)) * WIMG + px0 + (s & 7);
    Pack8 u;
#pragma unroll
    for (int j = 0; j < 8; ++j) u.h[j] = (__bf16)src[gp + (c0 + j) * HWPIX];
    *(uint4*)(sAB + s * SABS + c0) = u.v;
  }
}

__device__ __forceinline__ void win_gemm(const __bf16* sAB, const __bf16* wbf,
                                         const float* bias, float maskv,
                                         float* sQK, int tid) {
  int wv = tid >> 5, lane = tid & 31, hf = lane >> 4, lr = lane & 15;
  int mtile = wv >> 1, ntb = (wv & 1) * 4;
  f32x8 acc[4];
  for (int t = 0; t < 4; ++t) acc[t] = (f32x8)(0.0f);
#pragma unroll
  for (int kb = 0; kb < 128; kb += 32) {
    bf16x16 af = load_frag(sAB + (mtile * 16 + lr) * SABS + kb + hf * 8);
#pragma unroll
    for (int t = 0; t < 4; ++t) {
      bf16x16 bfr = load_frag(wbf + ((ntb + t) * 16 + lr) * 128 + kb + hf * 8);
      acc[t] = wmma_bf16(af, bfr, acc[t]);
    }
  }
#pragma unroll
  for (int t = 0; t < 4; ++t)
#pragma unroll
    for (int r = 0; r < 8; ++r) {
      int m = mtile * 16 + r + hf * 8, n = (ntb + t) * 16 + lr;
      sQK[m * SQKS + n] = maskv * acc[t][r] + bias[n];   // mask applies before bias
    }
}

__global__ __launch_bounds__(256) void CAMM_attn(
    const float* __restrict__ xin, const float* __restrict__ kimg, const float* __restrict__ v,
    const float* __restrict__ maskArr, const float* __restrict__ sa, const float* __restrict__ ca,
    const __bf16* __restrict__ qwbf, const float* __restrict__ qb,
    const __bf16* __restrict__ kwbf, const float* __restrict__ kbias,
    const float* __restrict__ ae1w, const float* __restrict__ bng, const float* __restrict__ bnb,
    const float* __restrict__ ae2w, const float* __restrict__ ae2b,
    float* __restrict__ outmid) {
  __shared__ alignas(16) float smem[15104];
  __bf16* sAB = (__bf16*)smem;          // 64*136 bf16 (4352 f)
  float* sQK = smem + 4352;             // 64*132
  float* sAg = smem + 4352 + 8448;      // 7*128
  float* sQA = sAg + 896;               // 448
  float* sAA = sQA + 448;               // 448
  float* sT1 = sAA + 448;               // 448
  float* sE2r = sT1 + 448;              // 49
  float* sE2b = sE2r + 49;              // 7

  int wid = blockIdx.x;
  int b = wid >> 10, win = wid & 1023;
  int py0 = (win >> 5) * 8, px0 = (win & 31) * 8;
  int xbase = b * 128 * HWPIX;
  int tid = threadIdx.x;
  float maskv = maskArr[wid];

  if (tid < 49) {  // 9-tap mean of ae2 folded into a 7x7 matrix
    int p = tid / 7, i = tid % 7;
    float s = 0.f;
    for (int j = 0; j < 9; ++j) s += ae2w[(p * 9 + j) * 7 + i];
    sE2r[tid] = s / 9.f;
  }
  if (tid < 7) {
    float s = 0.f;
    for (int j = 0; j < 9; ++j) s += ae2b[tid * 9 + j];
    sE2b[tid] = s / 9.f;
  }
  win_load_tile(sAB, xin, xbase, py0, px0, tid);
  __syncthreads();
  win_gemm(sAB, qwbf, qb, maskv, sQK, tid);             // q1
  __syncthreads();
  if (tid < 896) {                                      // agent = P @ q1
    int p = tid >> 7, c = tid & 127;
    int lo = (p * 64) / 7, hi = ((p + 1) * 64 + 6) / 7;
    float s = 0.f;
    for (int q = lo; q < hi; ++q) s += sQK[q * SQKS + c];
    sAg[tid] = s / (float)(hi - lo);
  }
  __syncthreads();
  if (tid < 448) {                                      // q_attn logits -> sQA[p*64+s]
    int s = tid / 7, p = tid % 7;
    float d = 0.f;
    for (int c = 0; c < 128; ++c) d += sQK[s * SQKS + c] * sAg[p * 128 + c];
    sQA[p * 64 + s] = d;
  }
  __syncthreads();
  if (tid < 64) {                                       // softmax over p -> q_attn^T
    float mx = -1e30f;
    for (int p = 0; p < 7; ++p) mx = fmaxf(mx, sQA[p * 64 + tid]);
    float sm = 0.f;
    for (int p = 0; p < 7; ++p) sm += expf(sQA[p * 64 + tid] - mx);
    for (int p = 0; p < 7; ++p) sQA[p * 64 + tid] = expf(sQA[p * 64 + tid] - mx) / sm;
  }
  __syncthreads();
  win_load_tile(sAB, kimg, xbase, py0, px0, tid);       // reload tile with k_img
  __syncthreads();
  win_gemm(sAB, kwbf, kbias, maskv, sQK, tid);          // k1 (q1 dead)
  __syncthreads();
  if (tid < 448) {                                      // agent_attn logits
    int p = tid / 64, s = tid % 64;
    float d = 0.f;
    for (int c = 0; c < 128; ++c) d += sAg[p * 128 + c] * sQK[s * SQKS + c];
    sAA[p * 64 + s] = d;
  }
  __syncthreads();
  if (tid < 7) {                                        // softmax over s
    float mx = -1e30f;
    for (int s = 0; s < 64; ++s) mx = fmaxf(mx, sAA[tid * 64 + s]);
    float sm = 0.f;
    for (int s = 0; s < 64; ++s) sm += expf(sAA[tid * 64 + s] - mx);
    float inv = 1.f / sm;
    for (int s = 0; s < 64; ++s) sAA[tid * 64 + s] = expf(sAA[tid * 64 + s] - mx) * inv;
  }
  __syncthreads();
  const float bnscale = 0.9999950000374997f;            // 1/sqrt(1+1e-5)
  if (tid < 448) {                                      // a1 = relu(bn(ae1 @ [aa;qaT]))
    int o = tid / 64, s = tid % 64;
    float d = 0.f;
#pragma unroll
    for (int i = 0; i < 14; ++i) {
      float yv = (i < 7) ? sAA[i * 64 + s] : sQA[(i - 7) * 64 + s];
      d += ae1w[o * 14 + i] * yv;
    }
    d = d * bnscale * bng[o] + bnb[o];
    sT1[tid] = d > 0.f ? d : 0.f;
  }
  __syncthreads();
  if (tid < 448) {                                      // m[p][s] (flat p-major)
    int p = tid / 64, s = tid % 64;
    float d = sE2b[p];
#pragma unroll
    for (int i = 0; i < 7; ++i) d += sE2r[p * 7 + i] * sT1[i * 64 + s];
    sAA[p * 64 + s] = d;
  }
  __syncthreads();
  if (tid < 64) {  // faithful raw reshape (7,64)->(64,7): row i = flat[i*7 .. i*7+7)
    float mx = -1e30f;
    for (int p = 0; p < 7; ++p) mx = fmaxf(mx, sAA[tid * 7 + p]);
    float sm = 0.f;
    for (int p = 0; p < 7; ++p) sm += expf(sAA[tid * 7 + p] - mx);
    float inv = 1.f / sm;
    for (int p = 0; p < 7; ++p) sQA[tid * 7 + p] = expf(sAA[tid * 7 + p] - mx) * inv;
  }
  __syncthreads();
  if (tid < 896) {                                      // v1p = P @ (mask*v_win)
    int p = tid >> 7, c = tid & 127;
    int lo = (p * 64) / 7, hi = ((p + 1) * 64 + 6) / 7;
    float s = 0.f;
    for (int q = lo; q < hi; ++q)
      s += v[xbase + c * HWPIX + (py0 + (q >> 3)) * WIMG + px0 + (q & 7)];
    sAg[tid] = maskv * s / (float)(hi - lo);
  }
  __syncthreads();
  float om = 1.f - maskv;
  for (int e = tid; e < 8192; e += 256) {               // k2 + k1 + (1-mask)*v*sa*ca
    int c = e >> 6, s = e & 63;
    int pix = (py0 + (s >> 3)) * WIMG + px0 + (s & 7);
    int gidx = xbase + c * HWPIX + pix;
    float d = sQK[s * SQKS + c];
#pragma unroll
    for (int p = 0; p < 7; ++p) d += sQA[s * 7 + p] * sAg[p * 128 + c];
    d += om * v[gidx] * sa[b * HWPIX + pix] * ca[b * 128 + c];
    outmid[gidx] = d;
  }
}

// ---------------------------------------------------------------------------
// Tail: depthwise convs + exact-GELU gate + residual
// ---------------------------------------------------------------------------
__global__ __launch_bounds__(256) void CAMM_dw3(const float* __restrict__ in,
                                                const float* __restrict__ w,
                                                const float* __restrict__ bias,
                                                float* __restrict__ out) {
  int idx = blockIdx.x * 256 + threadIdx.x;     // < 16777216
  int b = idx >> 23, rem = idx & 8388607;
  int c = rem >> 16, pix = rem & 65535;
  int y = pix >> 8, x = pix & 255;
  int base = b * 8388608 + c * HWPIX;
  float acc = bias[c];
  for (int ky = 0; ky < 3; ++ky) {
    int yy = y + ky - 1;
    if (yy < 0 || yy > 255) continue;
    for (int kx = 0; kx < 3; ++kx) {
      int xx = x + kx - 1;
      if (xx < 0 || xx > 255) continue;
      acc += in[base + yy * WIMG + xx] * w[c * 9 + ky * 3 + kx];
    }
  }
  out[idx] = acc;
}

__global__ __launch_bounds__(256) void CAMM_dw3d2_gelu(const float* __restrict__ in,
                                                       const float* __restrict__ w,
                                                       const float* __restrict__ bias,
                                                       const float* __restrict__ ca,
                                                       const float* __restrict__ resid,
                                                       float* __restrict__ out) {
  int idx = blockIdx.x * 256 + threadIdx.x;
  int b = idx >> 23, rem = idx & 8388607;
  int c = rem >> 16, pix = rem & 65535;
  int y = pix >> 8, x = pix & 255;
  int base = b * 8388608 + c * HWPIX;
  float acc = bias[c];
  for (int ky = 0; ky < 3; ++ky) {
    int yy = y + (ky - 1) * 2;
    if (yy < 0 || yy > 255) continue;
    for (int kx = 0; kx < 3; ++kx) {
      int xx = x + (kx - 1) * 2;
      if (xx < 0 || xx > 255) continue;
      acc += in[base + yy * WIMG + xx] * w[c * 9 + ky * 3 + kx];
    }
  }
  float g = 0.5f * acc * (1.f + erff(acc * 0.70710678118654752f));  // exact GELU
  out[idx] = g * ca[b * 128 + c] + resid[idx];
}

// ---------------------------------------------------------------------------
// Host launcher
// ---------------------------------------------------------------------------
extern "C" void kernel_launch(void* const* d_in, const int* in_sizes, int n_in,
                              void* d_out, int out_size, void* d_ws, size_t ws_size,
                              hipStream_t stream) {
  const float* x    = (const float*)d_in[0];
  const float* cg   = (const float*)d_in[1];
  const float* gn   = (const float*)d_in[2];
  const float* p_in_w = (const float*)d_in[3];
  const float* p_in_b = (const float*)d_in[4];
  const float* ln_g = (const float*)d_in[5];
  const float* ln_b = (const float*)d_in[6];
  const float* off1w = (const float*)d_in[7];
  const float* off1b = (const float*)d_in[8];
  const float* off2w = (const float*)d_in[9];
  const float* off2b = (const float*)d_in[10];
  const float* m1w = (const float*)d_in[11];
  const float* m1b = (const float*)d_in[12];
  const float* m2w = (const float*)d_in[13];
  const float* m2b = (const float*)d_in[14];
  const float* caw = (const float*)d_in[15];
  const float* cab = (const float*)d_in[16];
  const float* saw = (const float*)d_in[17];
  const float* sab = (const float*)d_in[18];
  const float* convw = (const float*)d_in[19];
  const float* convb = (const float*)d_in[20];
  const float* vw = (const float*)d_in[21];
  const float* vb = (const float*)d_in[22];
  const float* qw = (const float*)d_in[23];
  const float* qb = (const float*)d_in[24];
  const float* kw = (const float*)d_in[25];
  const float* kb = (const float*)d_in[26];
  const float* cs1w = (const float*)d_in[27];
  const float* cs1b = (const float*)d_in[28];
  const float* cs2w = (const float*)d_in[29];
  const float* cs2b = (const float*)d_in[30];
  const float* ow = (const float*)d_in[31];
  const float* ob = (const float*)d_in[32];
  const float* ae1w = (const float*)d_in[33];
  const float* bng = (const float*)d_in[34];
  const float* bnb = (const float*)d_in[35];
  const float* ae2w = (const float*)d_in[36];
  const float* ae2b = (const float*)d_in[37];

  float* ws = (float*)d_ws;
  float* V      = ws;                         // 16777216 (later reused as tail-in)
  float* KIMG   = ws + 16777216;              // 16777216 (later reused as y2)
  float* OUTMID = ws + 33554432;              // 16777216
  float* X2     = ws + 50331648;              // 4325376
  float* TMPIX  = ws + 54657024;              // 131072
  float* OFFX   = ws + 54788096;              // 131072
  float* OFFY   = ws + 54919168;              // 131072
  float* SA     = ws + 55050240;              // 131072
  float* PS     = ws + 55181312;              // 4096
  float* MASK   = ws + 55185408;              // 2048
  float* CA     = ws + 55187456;              // 256
  float* TMEAN  = ws + 55187712;              // 128
  float* PART   = ws + 55187840;              // 16896
  __bf16* BF    = (__bf16*)(ws + 55204736);   // 65536 bf16
  __bf16* qwbf = BF, *kwbf = BF + 16384, *vwbf = BF + 32768, *owbf = BF + 49152;

  CAMM_wcvt<<<256, 256, 0, stream>>>(qw, kw, vw, ow, BF);
  CAMM_gemm128<<<1024, 256, 0, stream>>>(x, vwbf, vb, V);
  CAMM_predictor<<<512, 256, 0, stream>>>(V, cg, p_in_w, p_in_b, ln_g, ln_b, convw, convb,
                                          off1w, off1b, off2w, off2b,
                                          TMPIX, X2, OFFX, OFFY, PART);
  CAMM_tmean_fin<<<1, 128, 0, stream>>>(PART, TMEAN);
  CAMM_ca<<<2, 128, 0, stream>>>(TMEAN, convw, convb, caw, cab, CA);
  CAMM_sa<<<512, 256, 0, stream>>>(X2, saw, sab, SA);
  CAMM_ps<<<64, 32, 0, stream>>>(TMPIX, m1w, m1b, m2w, m2b, PS);
  CAMM_mask<<<8, 256, 0, stream>>>(PS, gn, MASK);
  CAMM_warp<<<512, 256, 0, stream>>>(x, OFFX, OFFY, KIMG);
  CAMM_attn<<<2048, 256, 0, stream>>>(x, KIMG, V, MASK, SA, CA, qwbf, qb, kwbf, kb,
                                      ae1w, bng, bnb, ae2w, ae2b, OUTMID);
  CAMM_dw3<<<65536, 256, 0, stream>>>(OUTMID, cs1w, cs1b, KIMG);          // y2 into KIMG
  CAMM_dw3d2_gelu<<<65536, 256, 0, stream>>>(KIMG, cs2w, cs2b, CA, OUTMID, V); // tail-in into V
  CAMM_gemm128<<<1024, 256, 0, stream>>>(V, owbf, ob, (float*)d_out);
}
